// EdgePropertyPredictionModel_5360119185641
// MI455X (gfx1250) — matrix operand order repaired and checked by reading, not compiled, not run on hardware
//
#include <hip/hip_runtime.h>

#define N_NODES 40000
#define N_EDGES 640000
#define DIM     128
#define HEADS   8
#define DHEAD   16
#define LAYERS  3
#define IN_DIM  64
#define FF_DIM  218
#define FF_PAD  224
#define EPSV    1e-5f
#define SLOPE   0.2f

typedef __attribute__((ext_vector_type(16))) _Float16 v16h;
typedef __attribute__((ext_vector_type(8)))  float    v8f;

// ---------------- order-preserving float <-> uint for atomic max ------------
__device__ __forceinline__ unsigned fenc(float f) {
    unsigned u = __float_as_uint(f);
    return (u & 0x80000000u) ? ~u : (u | 0x80000000u);
}
__device__ __forceinline__ float fdec(unsigned e) {
    unsigned u = (e & 0x80000000u) ? (e & 0x7fffffffu) : ~e;
    return __uint_as_float(u);
}

// ---------------- generic fill ---------------------------------------------
__global__ void k_fill(unsigned* __restrict__ p, unsigned v, long n) {
    long i = (long)blockIdx.x * blockDim.x + threadIdx.x;
    long stride = (long)gridDim.x * blockDim.x;
    for (; i < n; i += stride) p[i] = v;
}

// ---------------- weight prep: transpose fp32 W[K x M] -> f16 Wt[Mpad x Kpad]
__global__ void k_prep_w(const float* __restrict__ W, _Float16* __restrict__ Wt,
                         int K, int M, int Kpad, int Mpad) {
    long i = (long)blockIdx.x * blockDim.x + threadIdx.x;
    long stride = (long)gridDim.x * blockDim.x;
    long tot = (long)Mpad * Kpad;
    for (; i < tot; i += stride) {
        int m = (int)(i / Kpad), k = (int)(i % Kpad);
        float v = (m < M && k < K) ? W[(long)k * M + m] : 0.f;
        Wt[i] = (_Float16)v;
    }
}

// ---------------- bias pad ---------------------------------------------------
__global__ void k_prep_bias(const float* __restrict__ b, float* __restrict__ bp,
                            int M, int Mpad) {
    int i = blockIdx.x * blockDim.x + threadIdx.x;
    if (i < Mpad) bp[i] = (i < M) ? b[i] : 0.f;
}

// ---------------- WMMA GEMM: C[N_NODES x Mpad] = A[N_NODES x K] * Wt^T ------
// Wt is f16, transposed+padded: Wt[m * K + k]. K multiple of 32, Mpad mult 32.
// Each wave: one 16-row x 32-col tile (2 accumulators sharing the A fragment).
__global__ void k_gemm(const float* __restrict__ A, const _Float16* __restrict__ Wt,
                       const float* __restrict__ bias, float* __restrict__ C,
                       int K, int ldc, int relu, int totalTiles, int colGroups) {
    int tile = blockIdx.x * 4 + threadIdx.y;      // 4 waves / block
    if (tile >= totalTiles) return;               // wave-uniform exit
    int rt = tile / colGroups;
    int cg = tile % colGroups;
    int lane = threadIdx.x;                       // 0..31 (wave32)
    int lrow = lane & 15;
    int lhi  = lane >> 4;
    long arow = (long)rt * 16 + lrow;
    int col0 = cg * 32 + lrow;
    int col1 = col0 + 16;

    const float*    Ap  = A  + arow * K;
    const _Float16* W0p = Wt + (long)col0 * K;
    const _Float16* W1p = Wt + (long)col1 * K;

    v8f acc0 = {0.f,0.f,0.f,0.f,0.f,0.f,0.f,0.f};
    v8f acc1 = {0.f,0.f,0.f,0.f,0.f,0.f,0.f,0.f};

    for (int k0 = 0; k0 < K; k0 += 32) {
        // A fragment: lanes 0-15 need k0+0..15; lanes 16-31: k0+16..31
        const float4* a0 = (const float4*)(Ap + k0 + lhi * 16);
        const float4* a1 = (const float4*)(Ap + k0 + 8 + lhi * 16);
        float4 x0 = a0[0], x1 = a0[1];
        float4 x2 = a1[0], x3 = a1[1];
        v16h af;
        af[0]  = (_Float16)x0.x; af[1]  = (_Float16)x0.y;
        af[2]  = (_Float16)x0.z; af[3]  = (_Float16)x0.w;
        af[4]  = (_Float16)x1.x; af[5]  = (_Float16)x1.y;
        af[6]  = (_Float16)x1.z; af[7]  = (_Float16)x1.w;
        af[8]  = (_Float16)x2.x; af[9]  = (_Float16)x2.y;
        af[10] = (_Float16)x2.z; af[11] = (_Float16)x2.w;
        af[12] = (_Float16)x3.x; af[13] = (_Float16)x3.y;
        af[14] = (_Float16)x3.z; af[15] = (_Float16)x3.w;
        // B fragments: 16 contiguous f16 per lane (lane = output col, K split by half)
        v16h bf0 = *(const v16h*)(W0p + k0 + lhi * 16);
        v16h bf1 = *(const v16h*)(W1p + k0 + lhi * 16);
        acc0 = __builtin_amdgcn_wmma_f32_16x16x32_f16(false, af, false, bf0,
                                                      (short)0, acc0, false, false);
        acc1 = __builtin_amdgcn_wmma_f32_16x16x32_f16(false, af, false, bf1,
                                                      (short)0, acc1, false, false);
    }

    float b0 = bias ? bias[col0] : 0.f;
    float b1 = bias ? bias[col1] : 0.f;
    #pragma unroll
    for (int v = 0; v < 8; ++v) {
        long row = (long)rt * 16 + v + 8 * lhi;   // C/D layout: VGPR v -> M = v+8*(lane/16)
        float y0 = acc0[v] + b0;
        float y1 = acc1[v] + b1;
        if (relu) { y0 = fmaxf(y0, 0.f); y1 = fmaxf(y1, 0.f); }
        C[row * ldc + col0] = y0;
        C[row * ldc + col1] = y1;
    }
}

// ---------------- attention logits el/er per (node, head) -------------------
__global__ void k_elr(const float* __restrict__ feat,
                      const float* __restrict__ al, const float* __restrict__ ar,
                      float* __restrict__ el, float* __restrict__ er) {
    long i = (long)blockIdx.x * blockDim.x + threadIdx.x;
    long stride = (long)gridDim.x * blockDim.x;
    for (; i < (long)N_NODES * HEADS; i += stride) {
        int n = (int)(i / HEADS), h = (int)(i % HEADS);
        const float* f = feat + (long)n * DIM + h * DHEAD;
        float sl = 0.f, sr = 0.f;
        #pragma unroll
        for (int d = 0; d < DHEAD; ++d) {
            sl += f[d] * al[h * DHEAD + d];
            sr += f[d] * ar[h * DHEAD + d];
        }
        el[i] = sl;
        er[i] = sr;
    }
}

// ---------------- edge pass 1: leaky-relu logits + segment max --------------
__global__ void k_edge1(const float* __restrict__ el, const float* __restrict__ er,
                        const int* __restrict__ src, const int* __restrict__ dst,
                        float* __restrict__ eBuf, unsigned* __restrict__ emaxE) {
    long i = (long)blockIdx.x * blockDim.x + threadIdx.x;
    long stride = (long)gridDim.x * blockDim.x;
    for (; i < N_EDGES; i += stride) {
        int s = src[i], d = dst[i];
        #pragma unroll
        for (int h = 0; h < HEADS; ++h) {
            float e = el[(long)s * HEADS + h] + er[(long)d * HEADS + h];
            e = (e > 0.f) ? e : SLOPE * e;
            eBuf[i * HEADS + h] = e;
            atomicMax(&emaxE[(long)d * HEADS + h], fenc(e));
        }
    }
}

// ---------------- edge pass 2: exp + segment sum ----------------------------
__global__ void k_edge2(const float* __restrict__ eBuf, const int* __restrict__ dst,
                        const unsigned* __restrict__ emaxE,
                        float* __restrict__ exBuf, float* __restrict__ den) {
    long i = (long)blockIdx.x * blockDim.x + threadIdx.x;
    long stride = (long)gridDim.x * blockDim.x;
    for (; i < N_EDGES; i += stride) {
        int d = dst[i];
        #pragma unroll
        for (int h = 0; h < HEADS; ++h) {
            float m = fdec(emaxE[(long)d * HEADS + h]);
            float ex = __expf(eBuf[i * HEADS + h] - m);
            exBuf[i * HEADS + h] = ex;
            atomicAdd(&den[(long)d * HEADS + h], ex);
        }
    }
}

// ---------------- edge pass 3: weighted message scatter ---------------------
__global__ void k_edge3(const float* __restrict__ exBuf, const float* __restrict__ den,
                        const float* __restrict__ feat,
                        const int* __restrict__ src, const int* __restrict__ dst,
                        float* __restrict__ msg) {
    long i = (long)blockIdx.x * blockDim.x + threadIdx.x;
    long stride = (long)gridDim.x * blockDim.x;
    for (; i < (long)N_EDGES * HEADS; i += stride) {
        long e = i / HEADS;
        int  h = (int)(i % HEADS);
        int  s = src[e], d = dst[e];
        float alpha = exBuf[i] / den[(long)d * HEADS + h];
        const float* f = feat + (long)s * DIM + h * DHEAD;
        float* o = msg + (long)d * DIM + h * DHEAD;
        #pragma unroll
        for (int dh = 0; dh < DHEAD; ++dh)
            atomicAdd(&o[dh], alpha * f[dh]);
    }
}

// ---------------- h += msg + bias(broadcast) --------------------------------
__global__ void k_add_msg(float* __restrict__ h, const float* __restrict__ msg,
                          const float* __restrict__ gb) {
    long i = (long)blockIdx.x * blockDim.x + threadIdx.x;
    long stride = (long)gridDim.x * blockDim.x;
    for (; i < (long)N_NODES * DIM; i += stride)
        h[i] += msg[i] + gb[i % DIM];
}

// ---------------- a += b ----------------------------------------------------
__global__ void k_add2(float* __restrict__ a, const float* __restrict__ b, long n) {
    long i = (long)blockIdx.x * blockDim.x + threadIdx.x;
    long stride = (long)gridDim.x * blockDim.x;
    for (; i < n; i += stride) a[i] += b[i];
}

// ---------------- BN column stats (sum, sumsq) over N rows, D=128 -----------
__global__ void k_bn_stats(const float* __restrict__ h, float* __restrict__ sums) {
    int col = threadIdx.x;                 // 128 threads = 128 cols
    float s = 0.f, s2 = 0.f;
    for (int r = blockIdx.x; r < N_NODES; r += gridDim.x) {
        float v = h[(long)r * DIM + col];
        s += v; s2 += v * v;
    }
    atomicAdd(&sums[col], s);
    atomicAdd(&sums[DIM + col], s2);
}

// ---------------- BN apply (optional relu) ----------------------------------
__global__ void k_bn_apply(float* __restrict__ h, const float* __restrict__ sums,
                           const float* __restrict__ g, const float* __restrict__ b,
                           int relu) {
    long i = (long)blockIdx.x * blockDim.x + threadIdx.x;
    long stride = (long)gridDim.x * blockDim.x;
    const float inv_n = 1.f / (float)N_NODES;
    for (; i < (long)N_NODES * DIM; i += stride) {
        int col = (int)(i % DIM);
        float mu  = sums[col] * inv_n;
        float var = sums[DIM + col] * inv_n - mu * mu;
        float y = g[col] * (h[i] - mu) * rsqrtf(var + EPSV) + b[col];
        if (relu) y = fmaxf(y, 0.f);
        h[i] = y;
    }
}

// ---------------- copy h into a 512-wide concat slice -----------------------
__global__ void k_copy_slice(float* __restrict__ hcat, const float* __restrict__ h, int off) {
    long i = (long)blockIdx.x * blockDim.x + threadIdx.x;
    long stride = (long)gridDim.x * blockDim.x;
    for (; i < (long)N_NODES * DIM; i += stride) {
        long n = i / DIM; int d = (int)(i % DIM);
        hcat[n * (4 * DIM) + off + d] = h[i];
    }
}

// ---------------- final 128 -> 1 projection ---------------------------------
__global__ void k_dot_out(const float* __restrict__ hdec, const float* __restrict__ w2,
                          float* __restrict__ out) {
    long i = (long)blockIdx.x * blockDim.x + threadIdx.x;
    long stride = (long)gridDim.x * blockDim.x;
    for (; i < N_NODES; i += stride) {
        const float* r = hdec + i * DIM;
        float s = 0.f;
        #pragma unroll
        for (int d = 0; d < DIM; ++d) s += r[d] * w2[d];
        out[i] = s;
    }
}

// ---------------- host-side helpers -----------------------------------------
static void gemm(hipStream_t st, const float* A, int K, const _Float16* Wt,
                 const float* bias, float* C, int Mpad, int relu) {
    int rowTiles = N_NODES / 16;          // 2500 (N_NODES multiple of 16)
    int colGroups = Mpad / 32;
    int tiles = rowTiles * colGroups;
    k_gemm<<<(tiles + 3) / 4, dim3(32, 4), 0, st>>>(A, Wt, bias, C, K, Mpad, relu,
                                                    tiles, colGroups);
}

static void prep_w(hipStream_t st, const float* W, _Float16* Wt,
                   int K, int M, int Kpad, int Mpad) {
    k_prep_w<<<256, 256, 0, st>>>(W, Wt, K, M, Kpad, Mpad);
}

static void fill_u32(hipStream_t st, void* p, unsigned v, long n) {
    long b = (n + 255) / 256;
    int blocks = (int)(b > 2048 ? 2048 : (b < 1 ? 1 : b));
    k_fill<<<blocks, 256, 0, st>>>((unsigned*)p, v, n);
}

extern "C" void kernel_launch(void* const* d_in, const int* in_sizes, int n_in,
                              void* d_out, int out_size, void* d_ws, size_t ws_size,
                              hipStream_t stream) {
    (void)in_sizes; (void)n_in; (void)out_size; (void)ws_size;
    const float* x      = (const float*)d_in[0];
    const int*   src    = (const int*)  d_in[1];
    const int*   dst    = (const int*)  d_in[2];
    const float* W_emb  = (const float*)d_in[3];
    const float* b_emb  = (const float*)d_in[4];
    const float* gat_W  = (const float*)d_in[5];
    const float* attn_l = (const float*)d_in[6];
    const float* attn_r = (const float*)d_in[7];
    const float* gat_b  = (const float*)d_in[8];
    const float* bn1_g  = (const float*)d_in[9];
    const float* bn1_b  = (const float*)d_in[10];
    const float* ff_W1  = (const float*)d_in[11];
    const float* ff_b1  = (const float*)d_in[12];
    const float* ff_W2  = (const float*)d_in[13];
    const float* ff_b2  = (const float*)d_in[14];
    const float* bn2_g  = (const float*)d_in[15];
    const float* bn2_b  = (const float*)d_in[16];
    const float* dec_W1 = (const float*)d_in[17];
    const float* dec_bn_g = (const float*)d_in[18];
    const float* dec_bn_b = (const float*)d_in[19];
    const float* dec_W2 = (const float*)d_in[20];
    float* out = (float*)d_out;

    // ---- carve workspace: fp32 region ----
    float* ws = (float*)d_ws;
    float*    h      = ws;                         ws += (long)N_NODES * DIM;
    float*    hcat   = ws;                         ws += (long)N_NODES * 4 * DIM;
    float*    feat   = ws;                         ws += (long)N_NODES * DIM;
    float*    msg    = ws;                         ws += (long)N_NODES * DIM;
    float*    y1     = ws;                         ws += (long)N_NODES * FF_PAD;
    float*    y2     = ws;                         ws += (long)N_NODES * DIM;
    float*    el     = ws;                         ws += (long)N_NODES * HEADS;
    float*    er     = ws;                         ws += (long)N_NODES * HEADS;
    float*    eBuf   = ws;                         ws += (long)N_EDGES * HEADS;
    float*    exBuf  = ws;                         ws += (long)N_EDGES * HEADS;
    unsigned* emaxE  = (unsigned*)ws;              ws += (long)N_NODES * HEADS;
    float*    den    = ws;                         ws += (long)N_NODES * HEADS;
    float*    bnsum  = ws;                         ws += 2 * DIM;
    float*    hdec   = ws;                         ws += (long)N_NODES * DIM;
    float*    fb1p   = ws;                         ws += LAYERS * FF_PAD;   // padded ff biases
    // ---- f16 region (32B-aligned: all counts above are multiples of 16) ----
    _Float16* hw = (_Float16*)ws;
    _Float16* Wemb_t = hw;                         hw += DIM * IN_DIM;            // [128][64]
    _Float16* gatW_t = hw;                         hw += (long)LAYERS * DIM * DIM;// [128][128] x3
    _Float16* ffW1_t = hw;                         hw += (long)LAYERS * FF_PAD * DIM; // [224][128] x3
    _Float16* ffW2_t = hw;                         hw += (long)LAYERS * DIM * FF_PAD; // [128][224] x3
    _Float16* decW_t = hw;                         hw += DIM * 4 * DIM;           // [128][512]

    const int EB = 2048, ET = 256;
    const long ND = (long)N_NODES * DIM;

    // ---- weight prep (transpose -> f16, pad FF dims) ----
    prep_w(stream, W_emb, Wemb_t, IN_DIM, DIM, IN_DIM, DIM);
    prep_w(stream, dec_W1, decW_t, 4 * DIM, DIM, 4 * DIM, DIM);
    for (int l = 0; l < LAYERS; ++l) {
        prep_w(stream, gat_W + (long)l * DIM * DIM, gatW_t + (long)l * DIM * DIM,
               DIM, DIM, DIM, DIM);
        prep_w(stream, ff_W1 + (long)l * DIM * FF_DIM, ffW1_t + (long)l * FF_PAD * DIM,
               DIM, FF_DIM, DIM, FF_PAD);
        prep_w(stream, ff_W2 + (long)l * FF_DIM * DIM, ffW2_t + (long)l * DIM * FF_PAD,
               FF_DIM, DIM, FF_PAD, DIM);
        k_prep_bias<<<1, FF_PAD, 0, stream>>>(ff_b1 + (long)l * FF_DIM,
                                              fb1p + (long)l * FF_PAD, FF_DIM, FF_PAD);
    }

    // ---- embed: h = x @ W_emb + b_emb ----
    gemm(stream, x, IN_DIM, Wemb_t, b_emb, h, DIM, 0);
    k_copy_slice<<<EB, ET, 0, stream>>>(hcat, h, 0);

    for (int l = 0; l < LAYERS; ++l) {
        // feat = h @ W_l
        gemm(stream, h, DIM, gatW_t + (long)l * DIM * DIM, nullptr, feat, DIM, 0);
        // attention logits
        k_elr<<<EB, ET, 0, stream>>>(feat, attn_l + (long)l * DIM, attn_r + (long)l * DIM,
                                     el, er);
        // reset segment buffers
        fill_u32(stream, emaxE, 0x007FFFFFu, (long)N_NODES * HEADS);  // enc(-inf)
        fill_u32(stream, den,   0u,          (long)N_NODES * HEADS);
        fill_u32(stream, msg,   0u,          ND);
        // segment softmax + scatter
        k_edge1<<<EB, ET, 0, stream>>>(el, er, src, dst, eBuf, emaxE);
        k_edge2<<<EB, ET, 0, stream>>>(eBuf, dst, emaxE, exBuf, den);
        k_edge3<<<EB, ET, 0, stream>>>(exBuf, den, feat, src, dst, msg);
        // skip connection + aggregation bias
        k_add_msg<<<EB, ET, 0, stream>>>(h, msg, gat_b + (long)l * DIM);
        // BN1
        fill_u32(stream, bnsum, 0u, 2 * DIM);
        k_bn_stats<<<256, DIM, 0, stream>>>(h, bnsum);
        k_bn_apply<<<EB, ET, 0, stream>>>(h, bnsum, bn1_g + (long)l * DIM,
                                          bn1_b + (long)l * DIM, 0);
        // FFN with skip (FF padded to 224; padded cols/rows are zero)
        gemm(stream, h, DIM, ffW1_t + (long)l * FF_PAD * DIM,
             fb1p + (long)l * FF_PAD, y1, FF_PAD, 1);
        gemm(stream, y1, FF_PAD, ffW2_t + (long)l * DIM * FF_PAD,
             ff_b2 + (long)l * DIM, y2, DIM, 0);
        k_add2<<<EB, ET, 0, stream>>>(h, y2, ND);
        // BN2
        fill_u32(stream, bnsum, 0u, 2 * DIM);
        k_bn_stats<<<256, DIM, 0, stream>>>(h, bnsum);
        k_bn_apply<<<EB, ET, 0, stream>>>(h, bnsum, bn2_g + (long)l * DIM,
                                          bn2_b + (long)l * DIM, 0);
        // JumpingKnowledge concat slice
        k_copy_slice<<<EB, ET, 0, stream>>>(hcat, h, (l + 1) * DIM);
    }

    // ---- decision MLP: relu(BN(hcat @ dec_W1)) @ dec_W2 ----
    gemm(stream, hcat, 4 * DIM, decW_t, nullptr, hdec, DIM, 0);
    fill_u32(stream, bnsum, 0u, 2 * DIM);
    k_bn_stats<<<256, DIM, 0, stream>>>(hdec, bnsum);
    k_bn_apply<<<EB, ET, 0, stream>>>(hdec, bnsum, dec_bn_g, dec_bn_b, 1);
    k_dot_out<<<EB, ET, 0, stream>>>(hdec, dec_W2, out);
}
